// PositionwiseAttention_23167053594794
// MI455X (gfx1250) — matrix operand order repaired and checked by reading, not compiled, not run on hardware
//
#include <hip/hip_runtime.h>
#include <hip/hip_bf16.h>

// ---------------------------------------------------------------------------
// PositionwiseAttention for MI455X (gfx1250, wave32, WMMA).
// B=32, C=KC=VC=N=1024. All five GEMMs are 1024^3 NT GEMMs on
// v_wmma_f32_16x16x32_bf16 with fp32 accumulation, double-buffered LDS fed by
// global_load_async_to_lds_b128 (ASYNCcnt pipeline). One-shot transposes make
// every B operand K-contiguous so staging is pure async b128 copies.
// ---------------------------------------------------------------------------

typedef __attribute__((ext_vector_type(16))) __bf16 v16bf;
typedef __attribute__((ext_vector_type(8)))  __bf16 v8bf;
typedef __attribute__((ext_vector_type(4)))  __bf16 v4bf;
typedef __attribute__((ext_vector_type(8)))  float  v8f;

constexpr int SZ  = 1024;   // M = N = K for every GEMM in this problem
constexpr int BM  = 128;    // block tile M
constexpr int BN  = 128;    // block tile N
constexpr int BK  = 32;     // K step (one WMMA K)
constexpr int LDK = 40;     // padded LDS row stride (80 B: 16B-aligned, bank-conflict-free)

// ---------------------------------------------------------------------------
// CDNA5 async global -> LDS copy (16 B per lane), tracked by ASYNCcnt.
// LDS byte address = low 32 bits of the flat pointer (ISA: LDS aperture
// truncates to addr[31:0], relative to the wave's LDS allocation).
// ---------------------------------------------------------------------------
__device__ __forceinline__ void async_copy_b128(const void* gsrc, void* ldst)
{
    unsigned           loff  = (unsigned)(unsigned long long)(size_t)ldst;
    unsigned long long gaddr = (unsigned long long)(size_t)gsrc;
    asm volatile("global_load_async_to_lds_b128 %0, %1, off"
                 :: "v"(loff), "v"(gaddr) : "memory");
}

// ---------------------------------------------------------------------------
// fp32 -> bf16 bulk conversion (weights)
// ---------------------------------------------------------------------------
__global__ __launch_bounds__(256) void cvt_f32_bf16_kernel(
    const float* __restrict__ in, __bf16* __restrict__ out, long n4)
{
    long i      = (long)blockIdx.x * blockDim.x + threadIdx.x;
    long stride = (long)gridDim.x * blockDim.x;
    for (; i < n4; i += stride) {
        float4 v = ((const float4*)in)[i];
        v4bf o;
        o.x = (__bf16)v.x; o.y = (__bf16)v.y; o.z = (__bf16)v.z; o.w = (__bf16)v.w;
        ((v4bf*)out)[i] = o;
    }
}

// ---------------------------------------------------------------------------
// Fused transpose + fp32->bf16:  out[b][n][c] = (bf16) in[b][c][n]
// 32x32 tiles via LDS. Grid: (32, 32, B), 256 threads.
// ---------------------------------------------------------------------------
__global__ __launch_bounds__(256) void transpose_cvt_f32_bf16_kernel(
    const float* __restrict__ in, __bf16* __restrict__ out)
{
    __shared__ float t[32][33];
    const int r0 = blockIdx.x * 32;            // source row block (c)
    const int c0 = blockIdx.y * 32;            // source col block (n)
    const int tx = threadIdx.x & 31;
    const int ty = threadIdx.x >> 5;           // 0..7
    const float* src = in  + (size_t)blockIdx.z * SZ * SZ;
    __bf16*      dst = out + (size_t)blockIdx.z * SZ * SZ;
#pragma unroll
    for (int r = 0; r < 4; ++r)
        t[ty + r * 8][tx] = src[(size_t)(r0 + ty + r * 8) * SZ + (c0 + tx)];
    __syncthreads();
#pragma unroll
    for (int r = 0; r < 4; ++r)
        dst[(size_t)(c0 + ty + r * 8) * SZ + (r0 + tx)] = (__bf16)t[tx][ty + r * 8];
}

// ---------------------------------------------------------------------------
// bf16 transpose: out[b][j][n] = in[b][n][j]. Grid: (32, 32, B), 256 threads.
// ---------------------------------------------------------------------------
__global__ __launch_bounds__(256) void transpose_bf16_kernel(
    const __bf16* __restrict__ in, __bf16* __restrict__ out)
{
    __shared__ __bf16 t[32][34];
    const int r0 = blockIdx.x * 32;
    const int c0 = blockIdx.y * 32;
    const int tx = threadIdx.x & 31;
    const int ty = threadIdx.x >> 5;
    const __bf16* src = in  + (size_t)blockIdx.z * SZ * SZ;
    __bf16*       dst = out + (size_t)blockIdx.z * SZ * SZ;
#pragma unroll
    for (int r = 0; r < 4; ++r)
        t[ty + r * 8][tx] = src[(size_t)(r0 + ty + r * 8) * SZ + (c0 + tx)];
    __syncthreads();
#pragma unroll
    for (int r = 0; r < 4; ++r)
        dst[(size_t)(c0 + ty + r * 8) * SZ + (r0 + tx)] = t[tx][ty + r * 8];
}

// ---------------------------------------------------------------------------
// NT bf16 WMMA GEMM:  C[m,n] = sum_k A[m,k] * B[n,k]
//   A: bf16 row-major [SZ x SZ] (K-contiguous);  B: bf16 [n-rows x k] (K-contiguous)
//   EPI 0: outBF[m,n] = bf16(acc + bias[m])
//   EPI 1: outF [m,n] = acc
//   EPI 2: outF [idx] = gamma[0]*acc + xres[idx]
// Grid: (SZ/BM, SZ/BN, batch), 256 threads (8 waves, 4x2; 32x64 per wave).
// Double-buffered LDS fed by async b128 copies (4 per thread per K-step).
// ---------------------------------------------------------------------------
template <int EPI>
__global__ __launch_bounds__(256) void gemm_bf16_wmma_kernel(
    const __bf16* __restrict__ A, const __bf16* __restrict__ B,
    size_t strideA, size_t strideB,
    const float* __restrict__ bias, const float* __restrict__ xres,
    const float* __restrict__ gamma,
    __bf16* __restrict__ outBF, float* __restrict__ outF, size_t strideC)
{
    __shared__ __bf16 As[2][BM * LDK];
    __shared__ __bf16 Bs[2][BN * LDK];

    const int batch = blockIdx.z;
    A += (size_t)batch * strideA;
    B += (size_t)batch * strideB;

    const int tid  = threadIdx.x;
    const int lane = tid & 31;
    const int wave = tid >> 5;
    const int wm   = wave >> 1;          // 0..3 -> 32-row strip
    const int wn   = wave & 1;           // 0..1 -> 64-col strip
    const int m0   = blockIdx.x * BM;
    const int n0   = blockIdx.y * BN;

    v8f zero = {0.f, 0.f, 0.f, 0.f, 0.f, 0.f, 0.f, 0.f};
    v8f acc[2][4];
#pragma unroll
    for (int i = 0; i < 2; ++i)
#pragma unroll
        for (int j = 0; j < 4; ++j) acc[i][j] = zero;

    // staging: thread t owns 32 B of the A tile and 32 B of the B tile
    const int arow = tid >> 1;            // 0..127
    const int acol = (tid & 1) * 16;      // 0 or 16 (bf16 elems)

    // fragment addressing (ISA 16-bit A / B operand layouts)
    const int fr = lane & 15;
    const int ka = (lane >> 4) * 8;       // A: K chunks {0-7,16-23} / {8-15,24-31}
    const int kb = (lane >> 4) * 8;       // B: K chunks {0-7,16-23} / {8-15,24-31}

    auto issue = [&](int buf, int k0) {
        const __bf16* ag = A + (size_t)(m0 + arow) * SZ + (k0 + acol);
        async_copy_b128(ag,     &As[buf][arow * LDK + acol]);
        async_copy_b128(ag + 8, &As[buf][arow * LDK + acol + 8]);
        const __bf16* bg = B + (size_t)(n0 + arow) * SZ + (k0 + acol);
        async_copy_b128(bg,     &Bs[buf][arow * LDK + acol]);
        async_copy_b128(bg + 8, &Bs[buf][arow * LDK + acol + 8]);
    };

    constexpr int NSTEP = SZ / BK;
    issue(0, 0);

    for (int i = 0; i < NSTEP; ++i) {
        const int cur = i & 1;
        if (i + 1 < NSTEP) {
            issue(cur ^ 1, (i + 1) * BK);
            // 4 newer copies outstanding; older 4 (current buffer) retired in order
            asm volatile("s_wait_asynccnt 0x4" ::: "memory");
        } else {
            asm volatile("s_wait_asynccnt 0x0" ::: "memory");
        }
        __syncthreads();   // current buffer fully resident for all waves

        v16bf aF[2], bF[4];
#pragma unroll
        for (int t = 0; t < 2; ++t) {
            const __bf16* p = &As[cur][(wm * 32 + t * 16 + fr) * LDK + ka];
            v8bf lo = *(const v8bf*)p;
            v8bf hi = *(const v8bf*)(p + 16);
            aF[t] = __builtin_shufflevector(lo, hi, 0,1,2,3,4,5,6,7,8,9,10,11,12,13,14,15);
        }
#pragma unroll
        for (int j = 0; j < 4; ++j) {
            const __bf16* p = &Bs[cur][(wn * 64 + j * 16 + fr) * LDK + kb];
            v8bf lo = *(const v8bf*)p;
            v8bf hi = *(const v8bf*)(p + 16);
            bF[j] = __builtin_shufflevector(lo, hi, 0,1,2,3,4,5,6,7,8,9,10,11,12,13,14,15);
        }
#pragma unroll
        for (int t = 0; t < 2; ++t)
#pragma unroll
            for (int j = 0; j < 4; ++j)
                acc[t][j] = __builtin_amdgcn_wmma_f32_16x16x32_bf16(
                    false, aF[t], false, bF[j], (short)0, acc[t][j], false, false);

        __syncthreads();   // all waves done reading buf[cur] before it is re-filled
    }

    // ---- epilogue -------------------------------------------------------
    const int rowoff = (lane >> 4) * 8;   // C/D layout: vgpr r -> m = r (+8 hi lanes)
    float g = 0.f;
    if constexpr (EPI == 2) g = gamma[0];

#pragma unroll
    for (int i = 0; i < 2; ++i) {
#pragma unroll
        for (int j = 0; j < 4; ++j) {
#pragma unroll
            for (int r = 0; r < 8; ++r) {
                const int m = m0 + wm * 32 + i * 16 + rowoff + r;
                const int n = n0 + wn * 64 + j * 16 + fr;
                const size_t idx = (size_t)batch * strideC + (size_t)m * SZ + n;
                const float v = acc[i][j][r];
                if constexpr (EPI == 0)      outBF[idx] = (__bf16)(v + bias[m]);
                else if constexpr (EPI == 1) outF[idx]  = v;
                else                         outF[idx]  = g * v + xres[idx];
            }
        }
    }
}

// ---------------------------------------------------------------------------
// Row softmax: fp32 logits [rows x 1024] -> bf16 attn. One block per row.
// ---------------------------------------------------------------------------
__global__ __launch_bounds__(256) void softmax_bf16_kernel(
    const float* __restrict__ logits, __bf16* __restrict__ attn)
{
    const size_t row = blockIdx.x;
    const float4* in = (const float4*)(logits + row * SZ);
    __bf16* out = attn + row * SZ;
    __shared__ float red[8];

    const int tid = threadIdx.x;
    float4 v = in[tid];

    float m = fmaxf(fmaxf(v.x, v.y), fmaxf(v.z, v.w));
#pragma unroll
    for (int off = 16; off > 0; off >>= 1) m = fmaxf(m, __shfl_xor(m, off, 32));
    if ((tid & 31) == 0) red[tid >> 5] = m;
    __syncthreads();
    float rm = red[0];
#pragma unroll
    for (int i = 1; i < 8; ++i) rm = fmaxf(rm, red[i]);
    __syncthreads();

    const float e0 = __expf(v.x - rm), e1 = __expf(v.y - rm);
    const float e2 = __expf(v.z - rm), e3 = __expf(v.w - rm);
    float s = (e0 + e1) + (e2 + e3);
#pragma unroll
    for (int off = 16; off > 0; off >>= 1) s += __shfl_xor(s, off, 32);
    if ((tid & 31) == 0) red[tid >> 5] = s;
    __syncthreads();
    float rs = red[0];
#pragma unroll
    for (int i = 1; i < 8; ++i) rs += red[i];
    const float inv = 1.0f / rs;

    v4bf o;
    o.x = (__bf16)(e0 * inv); o.y = (__bf16)(e1 * inv);
    o.z = (__bf16)(e2 * inv); o.w = (__bf16)(e3 * inv);
    ((v4bf*)out)[tid] = o;
}

// ---------------------------------------------------------------------------
// Host launch
// ---------------------------------------------------------------------------
extern "C" void kernel_launch(void* const* d_in, const int* in_sizes, int n_in,
                              void* d_out, int out_size, void* d_ws, size_t ws_size,
                              hipStream_t stream)
{
    constexpr int    Bt   = 32;
    constexpr size_t MAT  = (size_t)SZ * SZ;       // 1M elems per matrix
    constexpr size_t BMAT = (size_t)Bt * MAT;      // 32M elems per batched tensor

    const float* x     = (const float*)d_in[0];
    const float* Wq    = (const float*)d_in[1];
    const float* bq    = (const float*)d_in[2];
    const float* Wk    = (const float*)d_in[3];
    const float* bk    = (const float*)d_in[4];
    const float* Wv    = (const float*)d_in[5];
    const float* bv    = (const float*)d_in[6];
    const float* gamma = (const float*)d_in[7];

    char* p = (char*)d_ws;
    __bf16* xTb    = (__bf16*)p; p += BMAT * 2;    // x^T  [b][n][c] bf16
    __bf16* wqb    = (__bf16*)p; p += MAT * 2;
    __bf16* wkb    = (__bf16*)p; p += MAT * 2;
    __bf16* wvb    = (__bf16*)p; p += MAT * 2;
    __bf16* qb     = (__bf16*)p; p += BMAT * 2;    // q; reused as attn after stage 2
    __bf16* kb     = (__bf16*)p; p += BMAT * 2;    // k; reused as attn^T after stage 2
    __bf16* vb     = (__bf16*)p; p += BMAT * 2;
    float*  logits = (float*)p;  p += BMAT * 4;
    __bf16* attnb  = qb;                            // overlay (q dead after stage 2)
    __bf16* attnTb = kb;                            // overlay (k dead after stage 2)

    dim3 gt(SZ / 32, SZ / 32, Bt);
    transpose_cvt_f32_bf16_kernel<<<gt, 256, 0, stream>>>(x, xTb);
    cvt_f32_bf16_kernel<<<1024, 256, 0, stream>>>(Wq, wqb, (long)(MAT / 4));
    cvt_f32_bf16_kernel<<<1024, 256, 0, stream>>>(Wk, wkb, (long)(MAT / 4));
    cvt_f32_bf16_kernel<<<1024, 256, 0, stream>>>(Wv, wvb, (long)(MAT / 4));

    dim3 gg(SZ / BM, SZ / BN, Bt);

    // q/k/v projections: C[m,n] = sum_c W[m,c] * xT[n,c] + bias  (NT)
    gemm_bf16_wmma_kernel<0><<<gg, 256, 0, stream>>>(wqb, xTb, 0, MAT, bq, nullptr, nullptr, qb, nullptr, MAT);
    gemm_bf16_wmma_kernel<0><<<gg, 256, 0, stream>>>(wkb, xTb, 0, MAT, bk, nullptr, nullptr, kb, nullptr, MAT);
    gemm_bf16_wmma_kernel<0><<<gg, 256, 0, stream>>>(wvb, xTb, 0, MAT, bv, nullptr, nullptr, vb, nullptr, MAT);

    // logits[i,j] = sum_n q[i,n] * k[j,n]  (NT)
    gemm_bf16_wmma_kernel<1><<<gg, 256, 0, stream>>>(qb, kb, MAT, MAT, nullptr, nullptr, nullptr, nullptr, logits, MAT);

    // softmax rows -> bf16 attention, then transpose for the final NT GEMM
    softmax_bf16_kernel<<<Bt * SZ, 256, 0, stream>>>(logits, attnb);
    transpose_bf16_kernel<<<gt, 256, 0, stream>>>(attnb, attnTb);

    // out[v,j] = gamma * sum_n V[v,n] * attnT[j,n] + x  -> d_out (fp32)
    gemm_bf16_wmma_kernel<2><<<gg, 256, 0, stream>>>(vb, attnTb, MAT, MAT, nullptr, x, gamma, nullptr, (float*)d_out, MAT);
}
